// RelRepNoContext_53145925320930
// MI455X (gfx1250) — matrix operand order repaired and checked by reading, not compiled, not run on hardware
//
#include <hip/hip_runtime.h>

typedef _Float16 v16h __attribute__((ext_vector_type(16)));
typedef _Float16 v8h  __attribute__((ext_vector_type(8)));
typedef _Float16 v4h  __attribute__((ext_vector_type(4)));
typedef float    v8f  __attribute__((ext_vector_type(8)));
typedef float    v4f  __attribute__((ext_vector_type(4)));

#define HIDDEN    128
#define INPUT_DIM 256
#define INTER_DIM 384
#define NUM_SPANS 1024
#define NUM_RELS  16384
#define BATCH     8

#define W1_KSTEPS 8            // 256 / 32
#define W1_NTILES 24           // 384 / 16
#define W2_KSTEPS 12           // 384 / 32
#define W2_NTILES 8            // 128 / 16
#define W1_FRAGS  (W1_NTILES * W1_KSTEPS)   // 192
#define W2_FRAGS  (W2_NTILES * W2_KSTEPS)   // 96
#define FRAG_HALVES 512        // 32 lanes * 16 halves (1 KB)

#define MB 4                   // 16-row M-subtiles per wave (64 rows/wave)
#define WAVES_PER_BLOCK 2
#define ROWS_PER_WAVE (MB * 16)
#define TOTAL_ROWS (BATCH * NUM_RELS)                         // 131072
#define NBLOCKS (TOTAL_ROWS / (ROWS_PER_WAVE * WAVES_PER_BLOCK)) // 1024

// ---------------------------------------------------------------------------
// Prep: convert fp32 row-major W1/W2 into f16 WMMA B-fragment layout in d_ws.
// Fragment (nt, ks): lane L<16 holds column nt*16+L, K = ks*32 + {0..7,16..23};
// lane L>=16 holds column nt*16+(L-16), K = ks*32 + {8..15,24..31} (2 halves/VGPR).
// ---------------------------------------------------------------------------
__global__ __launch_bounds__(32) void prep_weights_kernel(
    const float* __restrict__ W1, const float* __restrict__ W2,
    _Float16* __restrict__ ws)
{
    const int f    = blockIdx.x;
    const int lane = threadIdx.x;

    const float* W;
    int ldn;
    _Float16* dst;
    int nt, ks;
    if (f < W1_FRAGS) {
        nt = f / W1_KSTEPS; ks = f % W1_KSTEPS;
        W = W1; ldn = INTER_DIM;
        dst = ws + (size_t)f * FRAG_HALVES;
    } else {
        const int g = f - W1_FRAGS;
        nt = g / W2_KSTEPS; ks = g % W2_KSTEPS;
        W = W2; ldn = HIDDEN;
        dst = ws + (size_t)W1_FRAGS * FRAG_HALVES + (size_t)g * FRAG_HALVES;
    }

    const int col   = nt * 16 + (lane & 15);
    const int kbase = ks * 32 + ((lane >> 4) << 3);

    v16h v;
#pragma unroll
    for (int j = 0; j < 16; ++j)
        v[j] = (_Float16)W[(size_t)(kbase + j + (j >= 8 ? 8 : 0)) * ldn + col];

    *(v16h*)(dst + lane * 16) = v;
}

// ---------------------------------------------------------------------------
// Main: each wave owns 64 relations (4 x 16-row subtiles).
// B fragments are held in VGPRs per N-tile and reused across the 4 subtiles;
// A fragments stream from LDS (ds_load_b128 pairs). Four independent
// accumulator chains per k-step give back-to-back WMMA issue.
// ---------------------------------------------------------------------------
__global__ __launch_bounds__(WAVES_PER_BLOCK * 32) void relrep_ffn_kernel(
    const float* __restrict__ span_reps,   // (B, S, H) fp32
    const int*   __restrict__ rel_ids,     // (B, R, 2) int32
    const float* __restrict__ b1,          // (384,)
    const float* __restrict__ b2,          // (128,)
    const _Float16* __restrict__ ws,       // prepped W1/W2 fragments
    float* __restrict__ out)               // (B, R, 128) fp32
{
    __shared__ __align__(16) _Float16 ldsX[WAVES_PER_BLOCK][ROWS_PER_WAVE * INPUT_DIM]; // 2*32KB
    __shared__ __align__(16) _Float16 ldsH[WAVES_PER_BLOCK][ROWS_PER_WAVE * INTER_DIM]; // 2*48KB

    const int tid  = threadIdx.x;
    const int wave = tid >> 5;
    const int lane = tid & 31;

    const int group = blockIdx.x * WAVES_PER_BLOCK + wave;  // 64-row group id
    const int row0  = group * ROWS_PER_WAVE;                // flattened (b, r) row
    const int b     = row0 >> 14;                           // / NUM_RELS
    const int r0    = row0 & (NUM_RELS - 1);

    const int2*  rids  = (const int2*)rel_ids + (size_t)b * NUM_RELS + r0;
    const float* spans = span_reps + (size_t)b * NUM_SPANS * HIDDEN;
    _Float16* X  = ldsX[wave];
    _Float16* Hs = ldsH[wave];

    // ---- Gather + fp32->f16 convert into LDS X tile (64 x 256 halves) ----
#pragma unroll 4
    for (int i = 0; i < ROWS_PER_WAVE; ++i) {
        const int2 ids = rids[i];                       // lane-uniform
        const v4f hv = ((const v4f*)(spans + (size_t)ids.x * HIDDEN))[lane];
        const v4f tv = ((const v4f*)(spans + (size_t)ids.y * HIDDEN))[lane];
        v4h hh, th;
#pragma unroll
        for (int j = 0; j < 4; ++j) { hh[j] = (_Float16)hv[j]; th[j] = (_Float16)tv[j]; }
        *(v4h*)(X + i * INPUT_DIM + lane * 4)          = hh;
        *(v4h*)(X + i * INPUT_DIM + HIDDEN + lane * 4) = th;
    }
    __syncthreads();

    const int row   = lane & 15;          // A-fragment row within subtile
    const int hiOff = (lane >> 4) << 3;   // +8 halves for upper half-wave
    const int mbase = (lane >> 4) << 3;   // C/D: VGPR v -> M = v (+8 upper lanes)
    const int ncol  = lane & 15;          // C/D column = lane % 16

    // ---- GEMM1: H = relu(X @ W1 + b1); B in registers, M from LDS ----
    for (int nt = 0; nt < W1_NTILES; ++nt) {
        v16h bf[W1_KSTEPS];
        const _Float16* wb = ws + (size_t)(nt * W1_KSTEPS) * FRAG_HALVES + lane * 16;
#pragma unroll
        for (int ks = 0; ks < W1_KSTEPS; ++ks)
            bf[ks] = *(const v16h*)(wb + ks * FRAG_HALVES);

        v8f acc[MB];
#pragma unroll
        for (int ms = 0; ms < MB; ++ms) acc[ms] = (v8f){};

#pragma unroll
        for (int ks = 0; ks < W1_KSTEPS; ++ks) {
#pragma unroll
            for (int ms = 0; ms < MB; ++ms) {
                const _Float16* p = X + (ms * 16 + row) * INPUT_DIM + ks * 32 + hiOff;
                const v8h lo = *(const v8h*)p;
                const v8h hi = *(const v8h*)(p + 16);
                const v16h a = __builtin_shufflevector(lo, hi,
                        0,1,2,3,4,5,6,7,8,9,10,11,12,13,14,15);
                acc[ms] = __builtin_amdgcn_wmma_f32_16x16x32_f16(
                        false, a, false, bf[ks], (short)0, acc[ms], false, false);
            }
        }

        const float bias = b1[nt * 16 + ncol];
        const int   col  = nt * 16 + ncol;
#pragma unroll
        for (int ms = 0; ms < MB; ++ms) {
#pragma unroll
            for (int v = 0; v < 8; ++v) {
                float h = acc[ms][v] + bias;
                h = h > 0.f ? h : 0.f;
                Hs[(ms * 16 + mbase + v) * INTER_DIM + col] = (_Float16)h;
            }
        }
    }
    __syncthreads();

    // ---- GEMM2: out = H @ W2 + b2; B in registers, M from LDS ----
    float* orow = out + (size_t)row0 * HIDDEN;
    const _Float16* w2base = ws + (size_t)W1_FRAGS * FRAG_HALVES;
    for (int nt = 0; nt < W2_NTILES; ++nt) {
        v16h bf[W2_KSTEPS];
        const _Float16* wb = w2base + (size_t)(nt * W2_KSTEPS) * FRAG_HALVES + lane * 16;
#pragma unroll
        for (int ks = 0; ks < W2_KSTEPS; ++ks)
            bf[ks] = *(const v16h*)(wb + ks * FRAG_HALVES);

        v8f acc[MB];
#pragma unroll
        for (int ms = 0; ms < MB; ++ms) acc[ms] = (v8f){};

#pragma unroll
        for (int ks = 0; ks < W2_KSTEPS; ++ks) {
#pragma unroll
            for (int ms = 0; ms < MB; ++ms) {
                const _Float16* p = Hs + (ms * 16 + row) * INTER_DIM + ks * 32 + hiOff;
                const v8h lo = *(const v8h*)p;
                const v8h hi = *(const v8h*)(p + 16);
                const v16h a = __builtin_shufflevector(lo, hi,
                        0,1,2,3,4,5,6,7,8,9,10,11,12,13,14,15);
                acc[ms] = __builtin_amdgcn_wmma_f32_16x16x32_f16(
                        false, a, false, bf[ks], (short)0, acc[ms], false, false);
            }
        }

        const float bias = b2[nt * 16 + ncol];
        const int   col  = nt * 16 + ncol;
#pragma unroll
        for (int ms = 0; ms < MB; ++ms) {
#pragma unroll
            for (int v = 0; v < 8; ++v)
                orow[(size_t)(ms * 16 + mbase + v) * HIDDEN + col] = acc[ms][v] + bias;
        }
    }
}

// ---------------------------------------------------------------------------
extern "C" void kernel_launch(void* const* d_in, const int* in_sizes, int n_in,
                              void* d_out, int out_size, void* d_ws, size_t ws_size,
                              hipStream_t stream) {
    const float* span_reps = (const float*)d_in[0];
    const int*   rel_ids   = (const int*)d_in[1];
    const float* W1        = (const float*)d_in[2];
    const float* b1        = (const float*)d_in[3];
    const float* W2        = (const float*)d_in[4];
    const float* b2        = (const float*)d_in[5];
    float*       out       = (float*)d_out;
    _Float16*    ws        = (_Float16*)d_ws;   // needs (192+96)*1KB = 288 KB

    prep_weights_kernel<<<W1_FRAGS + W2_FRAGS, 32, 0, stream>>>(W1, W2, ws);

    relrep_ffn_kernel<<<NBLOCKS, WAVES_PER_BLOCK * 32, 0, stream>>>(
        span_reps, rel_ids, b1, b2, ws, out);
}
